// SparseMemory_53240414601818
// MI455X (gfx1250) — matrix operand order repaired and checked by reading, not compiled, not run on hardware
//
#include <hip/hip_runtime.h>
#include <hip/hip_bf16.h>
#include <math.h>

// ---------------------------------------------------------------------------
// Sparse-DNC memory forward step for MI455X (gfx1250, wave32, WMMA).
// Only read_vectors (B,R,W) is returned by the reference, so all dead state
// updates (usage, post-link precedence, materialized link matrices, full
// memory scatter) are algebraically eliminated. The 128MB memory sweep — the
// only term that matters on the 23.3 TB/s roofline — is done once, fused:
// f16 WMMA (v_wmma_f32_16x16x32_f16) for q·m, VALU for ||m||^2.
// ---------------------------------------------------------------------------

typedef __attribute__((ext_vector_type(16))) _Float16 v16h;
typedef __attribute__((ext_vector_type(8)))  float    v8f;

namespace {

constexpr int RR  = 4;
constexpr int KK  = 10;
constexpr int KLL = 8;
constexpr int CC  = RR * KK + 2 * KLL + 1;   // 57
constexpr int BB  = 32;
constexpr int MM  = 16384;
constexpr int WW  = 64;
constexpr int INN = 512;
constexpr float EPSF    = 1e-6f;
constexpr float NEG_INF = -3.402823466e+38f;

__device__ __forceinline__ float sigmoidf_(float z) { return 1.f / (1.f + expf(-z)); }

// --- stable top-k insertion list: descending (value, then ascending index) --
template <int KN>
__device__ __forceinline__ void tk_init(float* v, int* ix) {
#pragma unroll
  for (int k = 0; k < KN; ++k) { v[k] = NEG_INF; ix[k] = 0x7FFFFFFF; }
}

template <int KN>
__device__ __forceinline__ void tk_insert(float* v, int* ix, float val, int idx) {
  if (val < v[KN - 1] || (val == v[KN - 1] && idx > ix[KN - 1])) return;
  int p = KN - 1;
  while (p > 0 && (val > v[p - 1] || (val == v[p - 1] && idx < ix[p - 1]))) {
    v[p] = v[p - 1]; ix[p] = ix[p - 1]; --p;
  }
  v[p] = val; ix[p] = idx;
}

// ---------------------------------------------------------------------------
// 1) Small projections: rq(B,R,W), wv(B,W), gate(B,C) sigmoid, wg(B) sigmoid.
// ---------------------------------------------------------------------------
__global__ __launch_bounds__(128)
void proj_kernel(const float* __restrict__ x,
                 const float* __restrict__ Wq, const float* __restrict__ bq,
                 const float* __restrict__ Wv, const float* __restrict__ bv,
                 const float* __restrict__ Wg, const float* __restrict__ bg,
                 const float* __restrict__ Wwg, const float* __restrict__ bwg,
                 float* __restrict__ rq, float* __restrict__ wv,
                 float* __restrict__ gate, float* __restrict__ wg) {
  const int b = blockIdx.x, t = threadIdx.x;
  __shared__ __align__(16) float xs[INN];
  for (int i = t; i < INN; i += 128) xs[i] = x[(size_t)b * INN + i];
  __syncthreads();
  constexpr int NOUT = RR * WW + WW + CC + 1;  // 378
  for (int o = t; o < NOUT; o += 128) {
    const float* wrow; float bias; int kind;
    if (o < RR * WW)                { wrow = Wq + (size_t)o * INN;                 bias = bq[o];                 kind = 0; }
    else if (o < RR * WW + WW)      { wrow = Wv + (size_t)(o - RR * WW) * INN;     bias = bv[o - RR * WW];      kind = 1; }
    else if (o < RR * WW + WW + CC) { wrow = Wg + (size_t)(o - RR * WW - WW) * INN; bias = bg[o - RR * WW - WW]; kind = 2; }
    else                            { wrow = Wwg;                                  bias = bwg[0];               kind = 3; }
    const float4* w4 = (const float4*)wrow;
    const float4* x4 = (const float4*)xs;
    float acc = 0.f;
#pragma unroll 4
    for (int i = 0; i < INN / 4; ++i) {
      const float4 a = w4[i], xb = x4[i];
      acc += a.x * xb.x + a.y * xb.y + a.z * xb.z + a.w * xb.w;
    }
    acc += bias;
    if (kind == 0)      rq[b * RR * WW + o] = acc;
    else if (kind == 1) wv[b * WW + (o - RR * WW)] = acc;
    else if (kind == 2) gate[b * CC + (o - RR * WW - WW)] = sigmoidf_(acc);
    else                wg[b] = sigmoidf_(acc);
  }
}

// ---------------------------------------------------------------------------
// 2) Copy write_weights into a mutable scratch (2MB) for the sparse scatter.
// ---------------------------------------------------------------------------
__global__ __launch_bounds__(256)
void copy_ww_kernel(const float* __restrict__ src, float* __restrict__ dst) {
  const size_t i = (size_t)blockIdx.x * 256 + threadIdx.x;
  dst[i] = src[i];
}

// ---------------------------------------------------------------------------
// 3) Per-batch sparse phase: Imin, ww_new, updated visible rows, last-flags,
//    least_used_mem update and the global clip bound.
// ---------------------------------------------------------------------------
__global__ __launch_bounds__(64)
void scatter_kernel(const int* __restrict__ pos, const float* __restrict__ read_weights,
                    const float* __restrict__ usage, const float* __restrict__ visible_memory,
                    const float* __restrict__ gates, const float* __restrict__ wgs,
                    const float* __restrict__ wvs, const int* __restrict__ lum,
                    float* __restrict__ ww_buf, float* __restrict__ vm_new,
                    int* __restrict__ lflag, int* __restrict__ lumn, int* __restrict__ cliph) {
  const int b = blockIdx.x, t = threadIdx.x;
  __shared__ int   p[CC];
  __shared__ float rel[CC], rwv[CC], wwn[CC], imin[CC];
  __shared__ float minu;
  if (t < CC) {
    p[t]   = pos[b * CC + t];
    rwv[t] = read_weights[(size_t)b * MM + p[t]];
    rel[t] = usage[(size_t)b * MM + p[t]];
  }
  __syncthreads();
  if (t == 0) {
    float mn = rel[0];
    for (int c = 1; c < CC; ++c) mn = fminf(mn, rel[c]);
    minu = mn;
  }
  __syncthreads();
  if (t < CC) {
    const float Im = (rel[t] == minu) ? 1.f : 0.f;
    const float g  = gates[b * CC + t];
    wwn[t]  = wgs[b] * (g * rwv[t] + (1.f - g) * Im);
    imin[t] = Im;
    int last = 1;
    for (int j = t + 1; j < CC; ++j)
      if (p[j] == p[t]) { last = 0; break; }
    lflag[b * CC + t] = last;
    if (last) ww_buf[(size_t)b * MM + p[t]] = wwn[t];  // duplicates: last wins
  }
  __syncthreads();
  for (int c = 0; c < CC; ++c)
    if (t < WW)
      vm_new[((size_t)b * CC + c) * WW + t] =
          visible_memory[((size_t)b * CC + c) * WW + t] * (1.f - imin[c]) +
          wwn[c] * wvs[b * WW + t];
  if (t == 0) {
    const int lum0  = lum[0];
    const int limit = (lum0 >= MM - 1) ? 1 : 0;
    lumn[b] = limit ? (CC + 1) : (lum[b] + 1);
    if (b == 0) cliph[0] = limit ? (CC + 1) : (lum0 + 1);
  }
}

// ---------------------------------------------------------------------------
// 4) Fused link update + fwd/bwd weights + top-8 each (never materializes the
//    updated (B,M,16) link matrices).
// ---------------------------------------------------------------------------
__global__ __launch_bounds__(256)
void link_topk_kernel(const float* __restrict__ lm, const float* __restrict__ rlm,
                      const float* __restrict__ prec, const float* __restrict__ read_weights,
                      const float* __restrict__ ww_buf, const int* __restrict__ pos,
                      int* __restrict__ fp, int* __restrict__ bp) {
  const int b = blockIdx.x, t = threadIdx.x;
  constexpr int L = 2 * KLL;  // 16
  __shared__ float trw[L], coef[L];
  __shared__ float s_pw, s_tw;
  __shared__ int   trp[L];
  if (t < L) {
    const int idx = pos[b * CC + (RR * KK + 1) + t];  // pos[:, 41:57]
    trp[t] = idx;
    trw[t] = read_weights[(size_t)b * MM + idx];
  }
  __syncthreads();
  if (t == 0) {
    float spw = 0.f, stw = 0.f;
    for (int l = 0; l < L; ++l) {
      const float tw = ww_buf[(size_t)b * MM + trp[l]];
      coef[l] = (1.f - tw) * trw[l];
      spw += prec[(size_t)b * MM + trp[l]] * trw[l];
      stw += tw * trw[l];
    }
    s_pw = spw; s_tw = stw;
  }
  __syncthreads();

  float fv[KLL]; int fi[KLL]; float bv_[KLL]; int bi_[KLL];
  tk_init<KLL>(fv, fi);
  tk_init<KLL>(bv_, bi_);
  for (int m = t; m < MM; m += 256) {
    const float4* l4 = (const float4*)(lm  + ((size_t)b * MM + m) * L);
    const float4* r4 = (const float4*)(rlm + ((size_t)b * MM + m) * L);
    float df = 0.f, db = 0.f;
#pragma unroll
    for (int i = 0; i < 4; ++i) {
      const float4 a = l4[i];
      df += a.x * trw[4 * i] + a.y * trw[4 * i + 1] + a.z * trw[4 * i + 2] + a.w * trw[4 * i + 3];
      const float4 c = r4[i];
      db += c.x * coef[4 * i] + c.y * coef[4 * i + 1] + c.z * coef[4 * i + 2] + c.w * coef[4 * i + 3];
    }
    const float wwm = ww_buf[(size_t)b * MM + m];
    tk_insert<KLL>(fv, fi, (1.f - wwm) * df + wwm * s_pw, m);
    tk_insert<KLL>(bv_, bi_, db + prec[(size_t)b * MM + m] * s_tw, m);
  }

  __shared__ float sv[256 * KLL];
  __shared__ int   si[256 * KLL];
  for (int k = 0; k < KLL; ++k) { sv[t * KLL + k] = fv[k]; si[t * KLL + k] = fi[k]; }
  __syncthreads();
  if (t == 0) {
    float mv[KLL]; int mi[KLL]; tk_init<KLL>(mv, mi);
    for (int e = 0; e < 256 * KLL; ++e) tk_insert<KLL>(mv, mi, sv[e], si[e]);
    for (int k = 0; k < KLL; ++k) fp[b * KLL + k] = mi[k];
  }
  __syncthreads();
  for (int k = 0; k < KLL; ++k) { sv[t * KLL + k] = bv_[k]; si[t * KLL + k] = bi_[k]; }
  __syncthreads();
  if (t == 0) {
    float mv[KLL]; int mi[KLL]; tk_init<KLL>(mv, mi);
    for (int e = 0; e < 256 * KLL; ++e) tk_insert<KLL>(mv, mi, sv[e], si[e]);
    for (int k = 0; k < KLL; ++k) bp[b * KLL + k] = mi[k];
  }
}

// ---------------------------------------------------------------------------
// 5) The 128MB memory sweep: d2[b,r,m] = ||mem||^2 - 2 q_r.mem via WMMA.
//    One wave = one 16-row tile. A = queries padded to 16 rows (f16),
//    B = memory tile (f16), K=64 split into two 16x16x32 WMMAs, f32 accum.
//    grid = (128, B), 8 waves/block -> 1024 tiles/batch, exactly M/16.
// ---------------------------------------------------------------------------
__global__ __launch_bounds__(256)
void d2_wmma_kernel(const float* __restrict__ memory,
                    const float* __restrict__ rq,
                    float* __restrict__ d2) {
  const int b    = blockIdx.y;
  const int wave = threadIdx.x >> 5;
  const int lane = threadIdx.x & 31;
  const int m15  = lane & 15;
  const int h    = lane >> 4;

  __shared__ __align__(16) float q[RR * WW];
  for (int i = threadIdx.x; i < RR * WW; i += 256) q[i] = rq[b * RR * WW + i];
  __syncthreads();

  // A fragments (16x32 f16): lane L<16 -> row M=L, K range [16h,16h+16).
  v16h a0, a1;
#pragma unroll
  for (int v = 0; v < 8; ++v) {
    const int kb = 2 * v + 16 * h;
    float f0 = 0.f, f1 = 0.f, g0 = 0.f, g1 = 0.f;
    if (m15 < RR) {
      f0 = q[m15 * WW + kb];      f1 = q[m15 * WW + kb + 1];
      g0 = q[m15 * WW + 32 + kb]; g1 = q[m15 * WW + 32 + kb + 1];
    }
    a0[2 * v] = (_Float16)f0; a0[2 * v + 1] = (_Float16)f1;
    a1[2 * v] = (_Float16)g0; a1[2 * v + 1] = (_Float16)g1;
  }

  const int tile = blockIdx.x * 8 + wave;             // [0, 1024)
  const int row  = tile * 16 + m15;
  const float* rp = memory + ((size_t)b * MM + row) * WW + 16 * h;
  __builtin_prefetch(rp, 0, 1);                       // global_prefetch_b8

  float4 xa[4], xb[4];
#pragma unroll
  for (int i = 0; i < 4; ++i) xa[i] = ((const float4*)rp)[i];
#pragma unroll
  for (int i = 0; i < 4; ++i) xb[i] = ((const float4*)(rp + 32))[i];

  float n2p = 0.f;
#pragma unroll
  for (int i = 0; i < 4; ++i) {
    n2p += xa[i].x * xa[i].x + xa[i].y * xa[i].y + xa[i].z * xa[i].z + xa[i].w * xa[i].w;
    n2p += xb[i].x * xb[i].x + xb[i].y * xb[i].y + xb[i].z * xb[i].z + xb[i].w * xb[i].w;
  }

  // B fragments (32x16 f16): lane holds column N=m15, K range [16h,16h+16).
  v16h fb0, fb1;
#pragma unroll
  for (int i = 0; i < 4; ++i) {
    fb0[4 * i + 0] = (_Float16)xa[i].x; fb0[4 * i + 1] = (_Float16)xa[i].y;
    fb0[4 * i + 2] = (_Float16)xa[i].z; fb0[4 * i + 3] = (_Float16)xa[i].w;
    fb1[4 * i + 0] = (_Float16)xb[i].x; fb1[4 * i + 1] = (_Float16)xb[i].y;
    fb1[4 * i + 2] = (_Float16)xb[i].z; fb1[4 * i + 3] = (_Float16)xb[i].w;
  }

  v8f c = {};
  c = __builtin_amdgcn_wmma_f32_16x16x32_f16(false, a0, false, fb0, (short)0, c, false, false);
  c = __builtin_amdgcn_wmma_f32_16x16x32_f16(false, a1, false, fb1, (short)0, c, false, false);

  const float n2 = n2p + __shfl_xor(n2p, 16);
  if (lane < 16) {
    // D layout: lanes 0-15 hold rows M=0..3 (queries) in c[0..3], N=lane.
    float* dp = d2 + ((size_t)b * RR) * MM + tile * 16 + lane;
#pragma unroll
    for (int r = 0; r < RR; ++r) dp[(size_t)r * MM] = n2 - 2.f * c[r];
  }
}

// ---------------------------------------------------------------------------
// 6) Fix up d2 for the <=57 rows whose memory row was rewritten this step.
// ---------------------------------------------------------------------------
__global__ __launch_bounds__(256)
void d2_fixup_kernel(const float* __restrict__ vm_new, const int* __restrict__ pos,
                     const int* __restrict__ lflag, const float* __restrict__ rq,
                     float* __restrict__ d2) {
  const int b = blockIdx.x, t = threadIdx.x;
  __shared__ float q[RR * WW];
  for (int i = t; i < RR * WW; i += 256) q[i] = rq[b * RR * WW + i];
  __syncthreads();
  for (int task = t; task < CC * RR; task += 256) {
    const int j = task >> 2, r = task & 3;
    if (!lflag[b * CC + j]) continue;
    const int row  = pos[b * CC + j];
    const float* v = vm_new + ((size_t)b * CC + j) * WW;
    float n2 = 0.f, s = 0.f;
    for (int w = 0; w < WW; ++w) {
      const float vv = v[w];
      n2 += vv * vv;
      s  += q[r * WW + w] * vv;
    }
    d2[((size_t)b * RR + r) * MM + row] = n2 - 2.f * s;
  }
}

// ---------------------------------------------------------------------------
// 7) Top-10 of -d2 per (b, r), lax.top_k-stable (ties -> lower index).
// ---------------------------------------------------------------------------
__global__ __launch_bounds__(256)
void knn_topk_kernel(const float* __restrict__ d2, int* __restrict__ knn) {
  const int br = blockIdx.x;  // b*RR + r
  const float* base = d2 + (size_t)br * MM;
  float tv[KK]; int ti[KK];
  tk_init<KK>(tv, ti);
  for (int m = threadIdx.x; m < MM; m += 256) tk_insert<KK>(tv, ti, -base[m], m);
  __shared__ float sv[256 * KK];
  __shared__ int   si[256 * KK];
  for (int k = 0; k < KK; ++k) { sv[threadIdx.x * KK + k] = tv[k]; si[threadIdx.x * KK + k] = ti[k]; }
  __syncthreads();
  if (threadIdx.x == 0) {
    float mv[KK]; int mi[KK]; tk_init<KK>(mv, mi);
    for (int e = 0; e < 256 * KK; ++e) tk_insert<KK>(mv, mi, sv[e], si[e]);
    const int b = br / RR, r = br % RR;
    for (int k = 0; k < KK; ++k) knn[b * (RR * KK) + r * KK + k] = mi[k];
  }
}

// ---------------------------------------------------------------------------
// 8) Finalize: new_pos (+clip), gather rows with override, cosine, softmax,
//    read_vectors -> d_out.
// ---------------------------------------------------------------------------
__global__ __launch_bounds__(128)
void finalize_kernel(const float* __restrict__ memory, const float* __restrict__ vm_new,
                     const int* __restrict__ pos, const int* __restrict__ lflag,
                     const int* __restrict__ knn, const int* __restrict__ fp,
                     const int* __restrict__ bp, const int* __restrict__ lumn,
                     const int* __restrict__ cliph, const float* __restrict__ rq,
                     float* __restrict__ out) {
  const int b = blockIdx.x, t = threadIdx.x;
  __shared__ int np[CC], p[CC], lf[CC];
  __shared__ __align__(16) float vm[CC][WW];   // 14.25 KB
  __shared__ float mn[CC], kn[RR];
  __shared__ float cosm[RR][CC];
  __shared__ __align__(16) float q[RR * WW];
  const int hi = cliph[0];
  if (t < CC) {
    int v;
    if (t < RR * KK)                 v = knn[b * (RR * KK) + t];
    else if (t < RR * KK + KLL)      v = fp[b * KLL + (t - RR * KK)];
    else if (t < RR * KK + 2 * KLL)  v = bp[b * KLL + (t - RR * KK - KLL)];
    else                             v = lumn[b];
    v = v < 0 ? 0 : (v > hi ? hi : v);
    np[t] = v;
    p[t]  = pos[b * CC + t];
    lf[t] = lflag[b * CC + t];
  }
  for (int i = t; i < RR * WW; i += 128) q[i] = rq[b * RR * WW + i];
  __syncthreads();
  for (int c = 0; c < CC; ++c) {
    const int row = np[c];
    int src = -1;
    for (int j = 0; j < CC; ++j)
      if (p[j] == row && lf[j]) src = j;   // last-write-wins row override
    const float* sp = (src >= 0) ? (vm_new + ((size_t)b * CC + src) * WW)
                                 : (memory + ((size_t)b * MM + row) * WW);
    if (t < WW) vm[c][t] = sp[t];
  }
  __syncthreads();
  if (t < CC) {
    float s = 0.f;
    for (int w = 0; w < WW; ++w) s += vm[t][w] * vm[t][w];
    mn[t] = sqrtf(s) + EPSF;
  } else if (t < CC + RR) {
    const int r = t - CC;
    float s = 0.f;
    for (int w = 0; w < WW; ++w) s += q[r * WW + w] * q[r * WW + w];
    kn[r] = sqrtf(s) + EPSF;
  }
  __syncthreads();
  for (int rc = t; rc < RR * CC; rc += 128) {
    const int r = rc / CC, c = rc % CC;
    float s = 0.f;
    for (int w = 0; w < WW; ++w) s += q[r * WW + w] * vm[c][w];
    cosm[r][c] = s / (kn[r] * mn[c]);
  }
  __syncthreads();
  if (t < RR) {
    float mx = cosm[t][0];
    for (int c = 1; c < CC; ++c) mx = fmaxf(mx, cosm[t][c]);
    float ssum = 0.f;
    for (int c = 0; c < CC; ++c) { const float e = expf(cosm[t][c] - mx); cosm[t][c] = e; ssum += e; }
    const float inv = 1.f / ssum;
    for (int c = 0; c < CC; ++c) cosm[t][c] *= inv;
  }
  __syncthreads();
  for (int rw = t; rw < RR * WW; rw += 128) {
    const int r = rw / WW, w = rw % WW;
    float s = 0.f;
    for (int c = 0; c < CC; ++c) s += cosm[r][c] * vm[c][w];
    out[(size_t)b * RR * WW + r * WW + w] = s;
  }
}

}  // anonymous namespace

extern "C" void kernel_launch(void* const* d_in, const int* in_sizes, int n_in,
                              void* d_out, int out_size, void* d_ws, size_t ws_size,
                              hipStream_t stream) {
  (void)in_sizes; (void)n_in; (void)out_size; (void)ws_size;
  const float* x               = (const float*)d_in[0];
  const float* memory          = (const float*)d_in[1];
  const float* visible_memory  = (const float*)d_in[2];
  const float* link_matrix     = (const float*)d_in[3];
  const float* rev_link_matrix = (const float*)d_in[4];
  const float* precedence      = (const float*)d_in[5];
  const float* read_weights    = (const float*)d_in[6];
  const float* write_weights   = (const float*)d_in[7];
  // d_in[8] usage is read; its update is dead w.r.t. the output.
  const float* usage           = (const float*)d_in[8];
  const float* Wq  = (const float*)d_in[9];
  const float* bq  = (const float*)d_in[10];
  const float* Wv  = (const float*)d_in[11];
  const float* bv  = (const float*)d_in[12];
  const float* Wg  = (const float*)d_in[13];
  const float* bg  = (const float*)d_in[14];
  const float* Wwg = (const float*)d_in[15];
  const float* bwg = (const float*)d_in[16];
  const int*   lum = (const int*)d_in[17];
  const int*   pos = (const int*)d_in[18];
  // d_in[19] timestep: only feeds the dead usage update.

  char* w = (char*)d_ws;
  auto alloc = [&](size_t bytes) -> char* {
    char* pch = w;
    w += (bytes + 255) & ~size_t(255);
    return pch;
  };
  float* ww_buf = (float*)alloc(sizeof(float) * BB * MM);        // 2 MB
  float* d2w    = (float*)alloc(sizeof(float) * BB * RR * MM);   // 8 MB
  float* rqs    = (float*)alloc(sizeof(float) * BB * RR * WW);
  float* wvs    = (float*)alloc(sizeof(float) * BB * WW);
  float* gates  = (float*)alloc(sizeof(float) * BB * CC);
  float* wgs    = (float*)alloc(sizeof(float) * BB);
  float* vm_new = (float*)alloc(sizeof(float) * BB * CC * WW);
  int*   knn    = (int*)alloc(sizeof(int) * BB * RR * KK);
  int*   fp     = (int*)alloc(sizeof(int) * BB * KLL);
  int*   bp     = (int*)alloc(sizeof(int) * BB * KLL);
  int*   lumn   = (int*)alloc(sizeof(int) * BB);
  int*   cliph  = (int*)alloc(sizeof(int) * 16);
  int*   lflag  = (int*)alloc(sizeof(int) * BB * CC);

  proj_kernel<<<BB, 128, 0, stream>>>(x, Wq, bq, Wv, bv, Wg, bg, Wwg, bwg,
                                      rqs, wvs, gates, wgs);
  copy_ww_kernel<<<(BB * MM) / 256, 256, 0, stream>>>(write_weights, ww_buf);
  scatter_kernel<<<BB, 64, 0, stream>>>(pos, read_weights, usage, visible_memory,
                                        gates, wgs, wvs, lum,
                                        ww_buf, vm_new, lflag, lumn, cliph);
  link_topk_kernel<<<BB, 256, 0, stream>>>(link_matrix, rev_link_matrix, precedence,
                                           read_weights, ww_buf, pos, fp, bp);
  d2_wmma_kernel<<<dim3(MM / 16 / 8, BB), 256, 0, stream>>>(memory, rqs, d2w);
  d2_fixup_kernel<<<BB, 256, 0, stream>>>(vm_new, pos, lflag, rqs, d2w);
  knn_topk_kernel<<<BB * RR, 256, 0, stream>>>(d2w, knn);
  finalize_kernel<<<BB, 128, 0, stream>>>(memory, vm_new, pos, lflag, knn, fp, bp,
                                          lumn, cliph, rqs, (float*)d_out);
}
